// GCN_44504451121629
// MI455X (gfx1250) — matrix operand order repaired and checked by reading, not compiled, not run on hardware
//
#include <hip/hip_runtime.h>
#include <cstdint>

// ---------------------------------------------------------------------------
// GCN link prediction for MI455X (gfx1250, wave32, WMMA).
// - Dense GEMMs: v_wmma_f32_16x16x32_bf16 (bf16 A/B, fp32 accumulate).
// - B operands (W1/W2/P1w) pre-packed ONCE to bf16 in the exact WMMA
//   B-fragment lane layout -> each fragment is one contiguous 32B load.
// - A fragments: fp32 global b128 loads + v_cvt_pk_bf16_f32, cached in
//   registers across column tiles.
// - Sparse aggregation: hardware global_atomic_add_f32 into the L2-resident
//   feature matrix.
// - Query head fully fused: gather*multiply -> WMMA -> bias/relu -> dot(P2w)
//   -> shfl reduce -> sigmoid. The 256->1 projection never touches memory.
// ---------------------------------------------------------------------------

typedef __attribute__((ext_vector_type(16))) __bf16 v16bf;
typedef __attribute__((ext_vector_type(8)))  float  v8f;

#define DIM      256
#define KSTEPS   8     // 256 / 32
#define COLTILES 16    // 256 / 16

// ---- WMMA fragment loaders -------------------------------------------------
// A (16x32 bf16) per-lane layout: lane L holds row M = L&15;
//   elements 0..7  = K = k0 + (L>>4)*8 + j
//   elements 8..15 = K = k0 + 16 + (L>>4)*8 + j
__device__ __forceinline__ v16bf a_frag_row(const float* __restrict__ rowptr,
                                            int k0, int lane) {
  const int b0 = (lane >> 4) << 3;  // 0 or 8
  const float4 x0 = *(const float4*)(rowptr + k0 + b0);
  const float4 x1 = *(const float4*)(rowptr + k0 + b0 + 4);
  const float4 x2 = *(const float4*)(rowptr + k0 + 16 + b0);
  const float4 x3 = *(const float4*)(rowptr + k0 + 16 + b0 + 4);
  v16bf a;
  a[0]  = (__bf16)x0.x; a[1]  = (__bf16)x0.y; a[2]  = (__bf16)x0.z; a[3]  = (__bf16)x0.w;
  a[4]  = (__bf16)x1.x; a[5]  = (__bf16)x1.y; a[6]  = (__bf16)x1.z; a[7]  = (__bf16)x1.w;
  a[8]  = (__bf16)x2.x; a[9]  = (__bf16)x2.y; a[10] = (__bf16)x2.z; a[11] = (__bf16)x2.w;
  a[12] = (__bf16)x3.x; a[13] = (__bf16)x3.y; a[14] = (__bf16)x3.z; a[15] = (__bf16)x3.w;
  return a;
}

// Packed-B fragment: 16 contiguous bf16 (32 bytes) per lane.
__device__ __forceinline__ v16bf b_frag_packed(const __bf16* __restrict__ Wp,
                                               int ct, int kb, int lane) {
  return *(const v16bf*)(Wp + ((((ct * KSTEPS) + kb) * 32 + lane) << 4));
}

// Pack W[256x256] fp32 (row-major K x N) into bf16 B-fragment layout:
//   Wp[((ct*8 + kb)*32 + lane)*16 + i] = W[kb*32 + (lane>>4)*8 + (i&7) + (i>>3)*16][ct*16 + (lane&15)]
__global__ void k_pack_w(const float* __restrict__ W, __bf16* __restrict__ Wp) {
  const int t = blockIdx.x * blockDim.x + threadIdx.x;  // 65536 threads
  if (t >= DIM * DIM) return;
  const int i    = t & 15;
  const int lane = (t >> 4) & 31;
  const int kb   = (t >> 9) & 7;
  const int ct   = t >> 12;
  const int K = kb * 32 + ((lane >> 4) << 3) + (i & 7) + ((i >> 3) << 4);
  const int n = ct * 16 + (lane & 15);
  Wp[t] = (__bf16)W[K * DIM + n];
}

// ---- Dense GEMM: C[M x 256] = A[M x 256] @ W[256 x 256] -------------------
// grid = (M/16, 4), block = 32 (one wave). A fragments cached in registers
// across 4 column tiles; B fragments are single 32B packed loads.
__global__ __launch_bounds__(32)
void k_gemm_bf16(const float* __restrict__ A, const __bf16* __restrict__ Wp,
                 float* __restrict__ C, int M) {
  const int lane = threadIdx.x;
  const int row0 = blockIdx.x * 16;
  if (row0 >= M) return;
  const float* rowptr = A + (size_t)(row0 + (lane & 15)) * DIM;

  v16bf afrag[KSTEPS];
#pragma unroll
  for (int kb = 0; kb < KSTEPS; ++kb)
    afrag[kb] = a_frag_row(rowptr, kb * 32, lane);

#pragma unroll
  for (int t = 0; t < 4; ++t) {
    const int ct = blockIdx.y * 4 + t;
    v8f acc = {};
#pragma unroll
    for (int kb = 0; kb < KSTEPS; ++kb) {
      v16bf b = b_frag_packed(Wp, ct, kb, lane);
      acc = __builtin_amdgcn_wmma_f32_16x16x32_bf16(
          false, afrag[kb], false, b, (short)0, acc, false, false);
    }
    // C layout: lane L, element i -> row row0 + (L>>4)*8 + i, col ct*16 + (L&15)
    float* cp = C + (size_t)(row0 + ((lane >> 4) << 3)) * DIM + ct * 16 + (lane & 15);
#pragma unroll
    for (int i = 0; i < 8; ++i) cp[(size_t)i * DIM] = acc[i];
  }
}

// ---- Fused query head ------------------------------------------------------
// grid = Q/16, block = 32. Per wave: 16 query rows.
//   h  = x[e0] * x[e1]            (gathered + multiplied into bf16 A frags)
//   y  = relu(h @ P1w + P1b)      (WMMA, 16 col tiles, packed B)
//   o  = sigmoid(y . P2w + P2b)   (fused in accumulator layout + shfl reduce)
__global__ __launch_bounds__(32)
void k_query(const float* __restrict__ X, const int* __restrict__ edges, int Q,
             const __bf16* __restrict__ P1wp, const float* __restrict__ P1b,
             const float* __restrict__ P2w, const float* __restrict__ P2b,
             float* __restrict__ out) {
  const int lane = threadIdx.x;
  const int q0   = blockIdx.x * 16;
  const int q    = q0 + (lane & 15);
  const float* ra = X + (size_t)edges[q] * DIM;
  const float* rb = X + (size_t)edges[Q + q] * DIM;

  v16bf afrag[KSTEPS];
  const int b0 = (lane >> 4) << 3;
#pragma unroll
  for (int kb = 0; kb < KSTEPS; ++kb) {
    const int base = kb * 32 + b0;
    const float4 a0 = *(const float4*)(ra + base);
    const float4 a1 = *(const float4*)(ra + base + 4);
    const float4 a2 = *(const float4*)(ra + base + 16);
    const float4 a3 = *(const float4*)(ra + base + 20);
    const float4 c0 = *(const float4*)(rb + base);
    const float4 c1 = *(const float4*)(rb + base + 4);
    const float4 c2 = *(const float4*)(rb + base + 16);
    const float4 c3 = *(const float4*)(rb + base + 20);
    v16bf a;
    a[0]  = (__bf16)(a0.x * c0.x); a[1]  = (__bf16)(a0.y * c0.y);
    a[2]  = (__bf16)(a0.z * c0.z); a[3]  = (__bf16)(a0.w * c0.w);
    a[4]  = (__bf16)(a1.x * c1.x); a[5]  = (__bf16)(a1.y * c1.y);
    a[6]  = (__bf16)(a1.z * c1.z); a[7]  = (__bf16)(a1.w * c1.w);
    a[8]  = (__bf16)(a2.x * c2.x); a[9]  = (__bf16)(a2.y * c2.y);
    a[10] = (__bf16)(a2.z * c2.z); a[11] = (__bf16)(a2.w * c2.w);
    a[12] = (__bf16)(a3.x * c3.x); a[13] = (__bf16)(a3.y * c3.y);
    a[14] = (__bf16)(a3.z * c3.z); a[15] = (__bf16)(a3.w * c3.w);
    afrag[kb] = a;
  }

  float s[8] = {0.f, 0.f, 0.f, 0.f, 0.f, 0.f, 0.f, 0.f};
#pragma unroll
  for (int ct = 0; ct < COLTILES; ++ct) {
    v8f acc = {};
#pragma unroll
    for (int kb = 0; kb < KSTEPS; ++kb) {
      v16bf b = b_frag_packed(P1wp, ct, kb, lane);
      acc = __builtin_amdgcn_wmma_f32_16x16x32_bf16(
          false, afrag[kb], false, b, (short)0, acc, false, false);
    }
    const int   col = ct * 16 + (lane & 15);
    const float bb  = P1b[col];
    const float w2  = P2w[col];
#pragma unroll
    for (int i = 0; i < 8; ++i) {
      float y = acc[i] + bb;
      s[i] += (y > 0.f ? y : 0.f) * w2;
    }
  }
  // Reduce over the 16-lane column group (masks stay inside the group).
#pragma unroll
  for (int i = 0; i < 8; ++i) {
    float v = s[i];
    v += __shfl_xor(v, 1, 32);
    v += __shfl_xor(v, 2, 32);
    v += __shfl_xor(v, 4, 32);
    v += __shfl_xor(v, 8, 32);
    s[i] = v;
  }
  if ((lane & 15) == 0) {
    const float b2 = P2b[0];
#pragma unroll
    for (int i = 0; i < 8; ++i) {
      const float z = s[i] + b2;
      out[q0 + ((lane >> 4) << 3) + i] = 1.0f / (1.0f + __expf(-z));
    }
  }
}

// ---- Sparse / elementwise helper kernels ----------------------------------
__global__ void k_fill(float* __restrict__ p, float v, int n) {
  int i = blockIdx.x * blockDim.x + threadIdx.x;
  if (i < n) p[i] = v;
}

__global__ void k_deg(const int* __restrict__ rows, float* __restrict__ deg, int E) {
  int i = blockIdx.x * blockDim.x + threadIdx.x;
  if (i < E) unsafeAtomicAdd(&deg[rows[i]], 1.0f);
}

__global__ void k_dinv(float* __restrict__ d, int n) {
  int i = blockIdx.x * blockDim.x + threadIdx.x;
  if (i < n) d[i] = rsqrtf(d[i]);  // deg >= 1 (self loops)
}

// x[i,:] = dinv[i]^2 * h[i,:]   (self-loop term, also zero-initializes x)
__global__ void k_selfinit(const float* __restrict__ h, const float* __restrict__ dinv,
                           float* __restrict__ x, int n) {
  int t = blockIdx.x * blockDim.x + threadIdx.x;
  if (t >= n * (DIM / 4)) return;
  const int i = t >> 6;
  const int d = (t & 63) << 2;
  float w = dinv[i]; w *= w;
  const float4 h4 = *(const float4*)(h + (size_t)i * DIM + d);
  float4 o;
  o.x = w * h4.x; o.y = w * h4.y; o.z = w * h4.z; o.w = w * h4.w;
  *(float4*)(x + (size_t)i * DIM + d) = o;
}

// x[row,:] += dinv[row]*dinv[col] * h[col,:] over all edges.
// 8 floats per thread: fewer redundant index/dinv loads; atomics to L2.
__global__ void k_scatter(const int* __restrict__ ei, const float* __restrict__ h,
                          const float* __restrict__ dinv, float* __restrict__ x, int E) {
  long long t = (long long)blockIdx.x * blockDim.x + threadIdx.x;
  if (t >= (long long)E * (DIM / 8)) return;
  const int e = (int)(t >> 5);
  const int d = ((int)t & 31) << 3;
  const int r = ei[e];
  const int c = ei[E + e];
  const float nm = dinv[r] * dinv[c];
  const float4 h4 = *(const float4*)(h + (size_t)c * DIM + d);
  const float4 h5 = *(const float4*)(h + (size_t)c * DIM + d + 4);
  float* xp = x + (size_t)r * DIM + d;
  unsafeAtomicAdd(xp + 0, nm * h4.x);
  unsafeAtomicAdd(xp + 1, nm * h4.y);
  unsafeAtomicAdd(xp + 2, nm * h4.z);
  unsafeAtomicAdd(xp + 3, nm * h4.w);
  unsafeAtomicAdd(xp + 4, nm * h5.x);
  unsafeAtomicAdd(xp + 5, nm * h5.y);
  unsafeAtomicAdd(xp + 6, nm * h5.z);
  unsafeAtomicAdd(xp + 7, nm * h5.w);
}

__global__ void k_bias_act(float* __restrict__ x, const float* __restrict__ b,
                           int n, int relu) {
  int t = blockIdx.x * blockDim.x + threadIdx.x;
  if (t >= n * (DIM / 4)) return;
  const int i = t >> 6;
  const int d = (t & 63) << 2;
  float4 v = *(float4*)(x + (size_t)i * DIM + d);
  const float4 bb = *(const float4*)(b + d);
  v.x += bb.x; v.y += bb.y; v.z += bb.z; v.w += bb.w;
  if (relu) {
    v.x = v.x > 0.f ? v.x : 0.f;
    v.y = v.y > 0.f ? v.y : 0.f;
    v.z = v.z > 0.f ? v.z : 0.f;
    v.w = v.w > 0.f ? v.w : 0.f;
  }
  *(float4*)(x + (size_t)i * DIM + d) = v;
}

// ---------------------------------------------------------------------------
extern "C" void kernel_launch(void* const* d_in, const int* in_sizes, int n_in,
                              void* d_out, int out_size, void* d_ws, size_t ws_size,
                              hipStream_t stream) {
  const int*   ei    = (const int*)d_in[0];    // [2, E]
  const int*   edges = (const int*)d_in[1];    // [2, Q]
  const float* emb   = (const float*)d_in[2];  // [N, 256]
  const float* W1    = (const float*)d_in[3];
  const float* b1    = (const float*)d_in[4];
  const float* W2    = (const float*)d_in[5];
  const float* b2    = (const float*)d_in[6];
  const float* P1w   = (const float*)d_in[7];
  const float* P1b   = (const float*)d_in[8];
  const float* P2w   = (const float*)d_in[9];
  const float* P2b   = (const float*)d_in[10];
  float* out = (float*)d_out;

  const int E = in_sizes[0] / 2;
  const int Q = in_sizes[1] / 2;
  const int N = in_sizes[2] / DIM;

  // Workspace: dinv [N pad] | W1p | W2p | P1wp (bf16, 64K elems each) | bufH | bufX
  float*  dinv = (float*)d_ws;
  __bf16* W1p  = (__bf16*)(dinv + ((N + 63) & ~63));
  __bf16* W2p  = W1p + DIM * DIM;
  __bf16* P1wp = W2p + DIM * DIM;
  float*  bufH = (float*)(P1wp + DIM * DIM);
  float*  bufX = bufH + (size_t)N * DIM;

  const int TB = 256;
  const int mtiles = (N + 15) / 16;       // N=50000 -> 3125 exact
  const int qtiles = (Q + 15) / 16;       // Q=200000 -> 12500 exact
  const int nd4 = N * (DIM / 4);
  const long long ed8 = (long long)E * (DIM / 8);
  const int packblocks = (DIM * DIM + TB - 1) / TB;

  // one-time weight packing (bf16, B-fragment layout)
  k_pack_w<<<packblocks, TB, 0, stream>>>(W1, W1p);
  k_pack_w<<<packblocks, TB, 0, stream>>>(W2, W2p);
  k_pack_w<<<packblocks, TB, 0, stream>>>(P1w, P1wp);

  // normalization weights
  k_fill<<<(N + TB - 1) / TB, TB, 0, stream>>>(dinv, 1.0f, N);
  k_deg<<<(E + TB - 1) / TB, TB, 0, stream>>>(ei, dinv, E);
  k_dinv<<<(N + TB - 1) / TB, TB, 0, stream>>>(dinv, N);

  // layer 1: h = emb @ W1 ; x = norm-aggregate(h) ; x = relu(x + b1)
  k_gemm_bf16<<<dim3(mtiles, 4), 32, 0, stream>>>(emb, W1p, bufH, N);
  k_selfinit<<<(nd4 + TB - 1) / TB, TB, 0, stream>>>(bufH, dinv, bufX, N);
  k_scatter<<<(unsigned)((ed8 + TB - 1) / TB), TB, 0, stream>>>(ei, bufH, dinv, bufX, E);
  k_bias_act<<<(nd4 + TB - 1) / TB, TB, 0, stream>>>(bufX, b1, N, 1);

  // layer 2: h = x @ W2 ; x = norm-aggregate(h) ; x = x + b2
  k_gemm_bf16<<<dim3(mtiles, 4), 32, 0, stream>>>(bufX, W2p, bufH, N);
  k_selfinit<<<(nd4 + TB - 1) / TB, TB, 0, stream>>>(bufH, dinv, bufX, N);
  k_scatter<<<(unsigned)((ed8 + TB - 1) / TB), TB, 0, stream>>>(ei, bufH, dinv, bufX, E);
  k_bias_act<<<(nd4 + TB - 1) / TB, TB, 0, stream>>>(bufX, b2, N, 0);

  // fused link-predictor head
  k_query<<<qtiles, 32, 0, stream>>>(bufX, edges, Q, P1wp, P1b, P2w, P2b, out);
}
